// ThreeBodyGravityLinearModel_76424648065233
// MI455X (gfx1250) — matrix operand order repaired and checked by reading, not compiled, not run on hardware
//
#include <hip/hip_runtime.h>
#include <hip/hip_bf16.h>

// ThreeBodyGravityLinearModel for MI455X (gfx1250, wave32, WMMA).
// Pair-MLP hidden GEMM [262144,256]@[256,256] x4 layers via v_wmma_f32_16x16x32_f16.
// Tiling: 256 rows/block (512 thr = 16 waves); each wave owns one N-tile and keeps
// its 8 B-fragments (W1 column block) in VGPRs for the whole GEMM -> W1 read once
// per block. A streams from LDS. Activations: hardware v_tanh_f32 (TRANS).

typedef __attribute__((ext_vector_type(16))) _Float16 v16h;
typedef __attribute__((ext_vector_type(8)))  _Float16 v8h;
typedef __attribute__((ext_vector_type(8)))  float    v8f;

#define H_DIM 256
#define KT    8          // 256 / 32 k-steps
#define ROWP  264        // padded f16 row stride (256+8): 528B = 33*16B, spreads LDS banks

#if __has_builtin(__builtin_amdgcn_tanhf)
__device__ __forceinline__ float fast_tanh(float x) { return __builtin_amdgcn_tanhf(x); }
#elif __has_builtin(__builtin_amdgcn_tanh_f32)
__device__ __forceinline__ float fast_tanh(float x) { return __builtin_amdgcn_tanh_f32(x); }
#else
__device__ __forceinline__ float fast_tanh(float x) { return tanhf(x); }
#endif

// pack two f32 -> one dword of f16 pair, single ds_store_b32
__device__ __forceinline__ void store_h16x2(_Float16* dst, float a, float b) {
  union { _Float16 h[2]; unsigned u; } pk;
  pk.h[0] = (_Float16)a;
  pk.h[1] = (_Float16)b;
  *(unsigned*)dst = pk.u;      // dst is 4B aligned (even col, even ROWP)
}

// ---- WMMA fragment loaders (layouts per CDNA5 ISA, wave32) ----
// A 16x32 f16: lane l: m=l&15, khi=l>>4; elems 0..7 -> k=kbase+8*khi+e ; 8..15 -> +16
__device__ __forceinline__ v16h ld_afrag(const _Float16* h, int row, int kbase, int khi) {
  const _Float16* base = h + row * ROWP + kbase + 8 * khi;  // 16B aligned
  v8h lo = *(const v8h*)(base);
  v8h hi = *(const v8h*)(base + 16);
  v16h a;
#pragma unroll
  for (int e = 0; e < 8; ++e) { a[e] = lo[e]; a[e + 8] = hi[e]; }
  return a;
}
// B 32x16 f16: lane l: n=l&15, khi=l>>4; elems 0..7 -> k = 8*khi + e ; 8..15 -> +16.
// Weights pre-swizzled so each lane's 16 halves are contiguous (32B): two b128 loads.
__device__ __forceinline__ v16h ld_bfrag(const _Float16* p) {
  v8h lo = *(const v8h*)(p);
  v8h hi = *(const v8h*)(p + 8);
  v16h b;
#pragma unroll
  for (int e = 0; e < 8; ++e) { b[e] = lo[e]; b[e + 8] = hi[e]; }
  return b;
}

// h2[256x256] = tanh(h1[256x256] @ W[256x256] + bias), f16 LDS -> f16 LDS.
// Wave owns N-tile nt == wave id (16 waves); B-frags hoisted in VGPRs; two M-tiles
// per iteration share each B-fragment (two independent accumulator chains).
__device__ __forceinline__ void gemm_hidden(const _Float16* __restrict__ h1,
                                            _Float16* __restrict__ h2,
                                            const _Float16* __restrict__ wt,
                                            const float* __restrict__ bias,
                                            int wave, int lane) {
  const int lm = lane & 15, khi = lane >> 4;
  const int nt = wave;                      // 16 waves == 16 N-tiles
  v16h bfr[KT];
#pragma unroll
  for (int kt = 0; kt < KT; ++kt)
    bfr[kt] = ld_bfrag(wt + ((size_t)(kt * 16 + nt) * 32 + lane) * 16);
  const float bv = bias[nt * 16 + lm];      // C/D col n = lane&15 -> bias folds into C
  for (int mt = 0; mt < 16; mt += 2) {
    v8f c0 = {bv, bv, bv, bv, bv, bv, bv, bv};
    v8f c1 = {bv, bv, bv, bv, bv, bv, bv, bv};
#pragma unroll
    for (int kt = 0; kt < KT; ++kt) {
      v16h a0 = ld_afrag(h1, (mt + 0) * 16 + lm, kt * 32, khi);
      v16h a1 = ld_afrag(h1, (mt + 1) * 16 + lm, kt * 32, khi);
      c0 = __builtin_amdgcn_wmma_f32_16x16x32_f16(false, a0, false, bfr[kt], (short)0, c0, false, false);
      c1 = __builtin_amdgcn_wmma_f32_16x16x32_f16(false, a1, false, bfr[kt], (short)0, c1, false, false);
    }
#pragma unroll
    for (int r = 0; r < 8; ++r) {           // C/D row m = 8*khi + r
      h2[((mt + 0) * 16 + 8 * khi + r) * ROWP + nt * 16 + lm] = (_Float16)fast_tanh(c0[r]);
      h2[((mt + 1) * 16 + 8 * khi + r) * ROWP + nt * 16 + lm] = (_Float16)fast_tanh(c1[r]);
    }
  }
}

// out[256 x ncols] = tanh(h2[256x256] @ W[256 x ncols<=16] + bias), B from LDS stage.
__device__ __forceinline__ void gemm_out(const _Float16* __restrict__ h2,
                                         const _Float16* __restrict__ wlds,
                                         const float* __restrict__ bias, int ncols,
                                         float* __restrict__ outg, int wave, int lane) {
  const int lm = lane & 15, khi = lane >> 4, mbase = wave * 16;   // 16 waves == 16 M-tiles
  const float bv = (lm < ncols) ? bias[lm] : 0.f;
  v8f c = {bv, bv, bv, bv, bv, bv, bv, bv};
#pragma unroll
  for (int kt = 0; kt < KT; ++kt) {
    v16h a = ld_afrag(h2, mbase + lm, kt * 32, khi);
    v16h b = ld_bfrag(wlds + ((size_t)kt * 32 + lane) * 16);
    c = __builtin_amdgcn_wmma_f32_16x16x32_f16(false, a, false, b, (short)0, c, false, false);
  }
  if (lm < ncols) {
#pragma unroll
    for (int r = 0; r < 8; ++r)
      outg[(size_t)(mbase + 8 * khi + r) * ncols + lm] = fast_tanh(c[r]);
  }
}

// ---- weight pre-swizzle: W[l][k][ncols] f32 -> tiles [l][kt][nt][lane][e] f16 ----
extern "C" __global__ __launch_bounds__(256)
void k_prep(const float* __restrict__ W, _Float16* __restrict__ out,
            int nLayers, int nTiles, int ncols) {
  int tid = blockIdx.x * 256 + threadIdx.x;
  int total = nLayers * KT * nTiles * 32;
  if (tid >= total) return;
  int lane = tid & 31;
  int r = tid >> 5;
  int nt = r % nTiles; r /= nTiles;
  int kt = r % KT;
  int l  = r / KT;
  int n  = nt * 16 + (lane & 15);
  int kb = kt * 32 + 8 * (lane >> 4);
  const float* w = W + (size_t)l * H_DIM * ncols;
  _Float16* o = out + (size_t)tid * 16;
#pragma unroll
  for (int e = 0; e < 16; ++e)
    o[e] = (n < ncols) ? (_Float16)w[(size_t)(kb + e + (e & 8)) * ncols + n] : (_Float16)0.f;
}

// ---- pair MLP: S entries s[b][l][p=i*16+j][9]; 256 points (one b,l) per block ----
extern "C" __global__ __launch_bounds__(512)
void k_pair_mlp(const float* __restrict__ m, const float* __restrict__ sW0,
                const float* __restrict__ sb0, const float* __restrict__ sb1,
                const float* __restrict__ sb2, const _Float16* __restrict__ w1t,
                const _Float16* __restrict__ w2t, float* __restrict__ s_out) {
  extern __shared__ char smem[];
  _Float16* h1 = (_Float16*)smem;               // 256*264 f16 = 135168 B
  _Float16* h2 = h1 + 256 * ROWP;               // 135168 B
  float* mv = (float*)(h2 + 256 * ROWP);        // 1024 B staging

  const int blk = blockIdx.x;                   // b*4 + l
  const int b = blk >> 2, l = blk & 3;
  const int t = threadIdx.x;

  if (t < 16) mv[t] = m[b * 16 + t];
  __syncthreads();

  // layer0: h1[p=i*16+j][h] = tanh(m[b][j]*W0[0][h] + m[b][i]*W0[1][h] + b0[h]).
  // Thread owns column pair (2c, 2c+1), c = t&127, point quarter g = t>>7 (64 pts).
  // m-values hoisted to registers (constant-index array); packed 2xf16 stores.
  {
    const int c2 = (t & 127) * 2;
    const int g  = t >> 7;
    const float w0a = sW0[l * 512 + c2],       w0b = sW0[l * 512 + c2 + 1];
    const float w1a = sW0[l * 512 + 256 + c2], w1b = sW0[l * 512 + 256 + c2 + 1];
    const float b0a = sb0[l * H_DIM + c2],     b0b = sb0[l * H_DIM + c2 + 1];
    float mj[16];
#pragma unroll
    for (int j = 0; j < 16; ++j) mj[j] = mv[j];
#pragma unroll
    for (int ii = 0; ii < 4; ++ii) {
      const float xi = mv[g * 4 + ii];          // i = p>>4, constant per 16 points
      const float ca = xi * w1a + b0a;
      const float cb = xi * w1b + b0b;
      _Float16* rowp = h1 + (g * 64 + ii * 16) * ROWP + c2;
#pragma unroll
      for (int j = 0; j < 16; ++j)
        store_h16x2(rowp + j * ROWP,
                    fast_tanh(mj[j] * w0a + ca),
                    fast_tanh(mj[j] * w0b + cb));
    }
  }
  __syncthreads();

  const int wave = t >> 5, lane = t & 31;
  gemm_hidden(h1, h2, w1t + (size_t)l * (KT * 16 * 32 * 16), sb1 + l * H_DIM, wave, lane);
  __syncthreads();

  // stage W2 layer tiles (8 KB) into the now-dead h1 buffer, then output GEMM
  {
    const unsigned* src = (const unsigned*)(w2t + (size_t)l * (KT * 32 * 16));
    unsigned* dst = (unsigned*)h1;
    for (int k = t; k < 2048; k += 512) dst[k] = src[k];
  }
  __syncthreads();
  float* sout = s_out + (size_t)(b * 4 + l) * 256 * 9;
  gemm_out(h2, (const _Float16*)h1, sb2 + l * 9, 9, sout, wave, lane);
}

// ---- bias MLPs (q / k): 16384 points, 256 per block ----
extern "C" __global__ __launch_bounds__(512)
void k_point_mlp(const float* __restrict__ m, const float* __restrict__ W0,
                 const float* __restrict__ b0, const float* __restrict__ b1,
                 const float* __restrict__ b2, const _Float16* __restrict__ w1t,
                 const _Float16* __restrict__ w2t, float* __restrict__ outg) {
  extern __shared__ char smem[];
  _Float16* h1 = (_Float16*)smem;
  _Float16* h2 = h1 + 256 * ROWP;
  float* mv = (float*)(h2 + 256 * ROWP);        // 256 floats

  const int pbase = blockIdx.x * 256;
  const int t = threadIdx.x;
  if (t < 256) mv[t] = m[pbase + t];
  __syncthreads();
  {
    const int c2 = (t & 127) * 2;
    const int g  = t >> 7;
    const float w0a = W0[c2], w0b = W0[c2 + 1];
    const float b0a = b0[c2], b0b = b0[c2 + 1];
    _Float16* rowp = h1 + (g * 64) * ROWP + c2;
    for (int pp = 0; pp < 64; ++pp) {
      const float x = mv[g * 64 + pp];
      store_h16x2(rowp + pp * ROWP,
                  fast_tanh(x * w0a + b0a),
                  fast_tanh(x * w0b + b0b));
    }
  }
  __syncthreads();
  const int wave = t >> 5, lane = t & 31;
  gemm_hidden(h1, h2, w1t, b1, wave, lane);
  __syncthreads();
  {
    const unsigned* src = (const unsigned*)w2t;
    unsigned* dst = (unsigned*)h1;
    for (int k = t; k < 2048; k += 512) dst[k] = src[k];
  }
  __syncthreads();
  gemm_out(h2, (const _Float16*)h1, b2, 3, outg + (size_t)pbase * 3, wave, lane);
}

// ---- apply chain: per batch, Sfull = S + S^T (block 48x48), alternate p/q updates ----
extern "C" __global__ __launch_bounds__(64)
void k_apply(const float* __restrict__ q, const float* __restrict__ p,
             const float* __restrict__ dtp, const float* __restrict__ s_all,
             const float* __restrict__ bq, const float* __restrict__ bp,
             float* __restrict__ out) {
  __shared__ float xq[48], xp[48], sbuf[2304];
  const int b = blockIdx.x, t = threadIdx.x;
  if (t < 48) { xq[t] = q[b * 48 + t]; xp[t] = p[b * 48 + t]; }
  const float dt = dtp[0];
  const float scale = dt * dt * dt;
  for (int l = 0; l < 4; ++l) {
    const float* sl = s_all + (size_t)(b * 4 + l) * 2304;
    for (int k = t; k < 2304; k += 64) sbuf[k] = sl[k];
    __syncthreads();
    if (t < 48) {
      const int i = t / 3, a = t - 3 * i;
      const float* xs = (l & 1) ? xp : xq;
      float acc = 0.f;
      for (int j = 0; j < 16; ++j) {
        const float* s1 = sbuf + (i * 16 + j) * 9 + a * 3;   // S[i,j][a,c]
        const float* s2 = sbuf + (j * 16 + i) * 9 + a;       // S[j,i][c,a]
#pragma unroll
        for (int c = 0; c < 3; ++c)
          acc += (s1[c] + s2[c * 3]) * xs[j * 3 + c];
      }
      if (l & 1) xq[t] += acc * scale; else xp[t] += acc * scale;
    }
    __syncthreads();
  }
  if (t < 48) {
    out[(size_t)b * 48 + t]          = xq[t] + bq[b * 48 + t] * scale;
    out[49152 + (size_t)b * 48 + t]  = xp[t] + bp[b * 48 + t] * scale;
  }
}

extern "C" void kernel_launch(void* const* d_in, const int* in_sizes, int n_in,
                              void* d_out, int out_size, void* d_ws, size_t ws_size,
                              hipStream_t stream) {
  (void)in_sizes; (void)n_in; (void)out_size; (void)ws_size;
  const float* q   = (const float*)d_in[0];
  const float* p   = (const float*)d_in[1];
  const float* m   = (const float*)d_in[2];
  const float* dt  = (const float*)d_in[3];
  const float* sW0 = (const float*)d_in[4];
  const float* sb0 = (const float*)d_in[5];
  const float* sW1 = (const float*)d_in[6];
  const float* sb1 = (const float*)d_in[7];
  const float* sW2 = (const float*)d_in[8];
  const float* sb2 = (const float*)d_in[9];
  const float* qW0 = (const float*)d_in[10];
  const float* qb0 = (const float*)d_in[11];
  const float* qW1 = (const float*)d_in[12];
  const float* qb1 = (const float*)d_in[13];
  const float* qW2 = (const float*)d_in[14];
  const float* qb2 = (const float*)d_in[15];
  const float* kW0 = (const float*)d_in[16];
  const float* kb0 = (const float*)d_in[17];
  const float* kW1 = (const float*)d_in[18];
  const float* kb1 = (const float*)d_in[19];
  const float* kW2 = (const float*)d_in[20];
  const float* kb2 = (const float*)d_in[21];

  char* ws = (char*)d_ws;
  const size_t OFF_S    = 0;                                        // 1024*4*256*9 f32
  const size_t OFF_BQ   = OFF_S    + (size_t)1024 * 4 * 256 * 9 * 4;
  const size_t OFF_BP   = OFF_BQ   + (size_t)49152 * 4;
  const size_t OFF_W1T  = OFF_BP   + (size_t)49152 * 4;             // 4 layers B-tiles f16
  const size_t OFF_W2T  = OFF_W1T  + (size_t)4 * KT * 16 * 32 * 16 * 2;
  const size_t OFF_QW1T = OFF_W2T  + (size_t)4 * KT * 1 * 32 * 16 * 2;
  const size_t OFF_KW1T = OFF_QW1T + (size_t)KT * 16 * 32 * 16 * 2;
  const size_t OFF_QW2T = OFF_KW1T + (size_t)KT * 16 * 32 * 16 * 2;
  const size_t OFF_KW2T = OFF_QW2T + (size_t)KT * 32 * 16 * 2;

  float*    s_all = (float*)(ws + OFF_S);
  float*    bqg   = (float*)(ws + OFF_BQ);
  float*    bpg   = (float*)(ws + OFF_BP);
  _Float16* w1t   = (_Float16*)(ws + OFF_W1T);
  _Float16* w2t   = (_Float16*)(ws + OFF_W2T);
  _Float16* qw1t  = (_Float16*)(ws + OFF_QW1T);
  _Float16* kw1t  = (_Float16*)(ws + OFF_KW1T);
  _Float16* qw2t  = (_Float16*)(ws + OFF_QW2T);
  _Float16* kw2t  = (_Float16*)(ws + OFF_KW2T);

  // 1) pre-swizzle weights to f16 WMMA B-fragment tiles
  k_prep<<<64, 256, 0, stream>>>(sW1, w1t, 4, 16, 256);
  k_prep<<<4,  256, 0, stream>>>(sW2, w2t, 4, 1, 9);
  k_prep<<<16, 256, 0, stream>>>(qW1, qw1t, 1, 16, 256);
  k_prep<<<1,  256, 0, stream>>>(qW2, qw2t, 1, 1, 3);
  k_prep<<<16, 256, 0, stream>>>(kW1, kw1t, 1, 16, 256);
  k_prep<<<1,  256, 0, stream>>>(kW2, kw2t, 1, 1, 3);

  // 2) big WMMA passes (dynamic LDS: two 256x264 f16 buffers + staging)
  const size_t smem = (size_t)2 * 256 * ROWP * 2 + 1024;  // 271360 B -> 1 WG / 320KB WGP
  k_pair_mlp<<<1024 * 4, 512, smem, stream>>>(m, sW0, sb0, sb1, sb2, w1t, w2t, s_all);
  k_point_mlp<<<64, 512, smem, stream>>>(m, qW0, qb0, qb1, qb2, qw1t, qw2t, bqg);
  k_point_mlp<<<64, 512, smem, stream>>>(m, kW0, kb0, kb1, kb2, kw1t, kw2t, bpg);

  // 3) symmetrized 48x48 chain + bias
  k_apply<<<1024, 64, 0, stream>>>(q, p, dt, s_all, bqg, bpg, (float*)d_out);
}
